// SwinBlock_59931973648812
// MI455X (gfx1250) — compile-verified, matrix operand order checked by reading, not run on hardware
//
#include <hip/hip_runtime.h>
#include <hip/hip_bf16.h>
#include <math.h>

typedef __attribute__((ext_vector_type(16))) _Float16 v16h;
typedef __attribute__((ext_vector_type(8)))  _Float16 v8h;
typedef __attribute__((ext_vector_type(8)))  float    v8f;

#define NTOK 49
#define CC   128
#define NHEAD 4
#define HID  512
#define HW   56
#define SHFT 3
#define LN_EPS 1e-5f

#define WMMA_F16(a, b, c) \
  __builtin_amdgcn_wmma_f32_16x16x32_f16(false, (a), false, (b), (short)0, (c), false, false)

__device__ __forceinline__ float wred_sum(float v) {
#pragma unroll
  for (int o = 16; o > 0; o >>= 1) v += __shfl_xor(v, o, 32);
  return v;
}

// A/B-operand fragment (16x32 f16 tile, rows striped over lanes, K along VGPRs).
// Per ISA 7.12.2 the per-lane halves are contiguous in two 16B runs: K+[0..7] and K+[16..23]
// (K offset +8 for lanes 16..31) -> two 128-bit loads.
__device__ __forceinline__ v16h ld_a(const _Float16* buf, int ld, int row0, int col0, int lane) {
  int m  = row0 + (lane & 15);
  int kb = (lane >> 4) << 3;
  const v8h* p = (const v8h*)(buf + m * ld + col0 + kb);
  v8h lo = p[0];
  v8h hi = p[2];
  return __builtin_shufflevector(lo, hi, 0, 1, 2, 3, 4, 5, 6, 7,
                                         8, 9, 10, 11, 12, 13, 14, 15);
}

// Same fragment sourced from an f32 buffer (two 32B runs), converting to f16.
__device__ __forceinline__ v16h ld_a_f32(const float* buf, int ld, int row0, int col0, int lane) {
  int m  = row0 + (lane & 15);
  int kb = (lane >> 4) << 3;
  const float4* p = (const float4*)(buf + m * ld + col0 + kb);
  float4 q0 = p[0], q1 = p[1], q2 = p[4], q3 = p[5];
  v16h a;
  a[0]  = (_Float16)q0.x; a[1]  = (_Float16)q0.y; a[2]  = (_Float16)q0.z; a[3]  = (_Float16)q0.w;
  a[4]  = (_Float16)q1.x; a[5]  = (_Float16)q1.y; a[6]  = (_Float16)q1.z; a[7]  = (_Float16)q1.w;
  a[8]  = (_Float16)q2.x; a[9]  = (_Float16)q2.y; a[10] = (_Float16)q2.z; a[11] = (_Float16)q2.w;
  a[12] = (_Float16)q3.x; a[13] = (_Float16)q3.y; a[14] = (_Float16)q3.z; a[15] = (_Float16)q3.w;
  return a;
}

// ---- weight prep: wt[n][k] = (f16) w[k][n] -----------------------------------
__global__ void k_wprep(const float* __restrict__ w, _Float16* __restrict__ wt, int K, int N) {
  int total = K * N;
  for (int i = blockIdx.x * blockDim.x + threadIdx.x; i < total; i += gridDim.x * blockDim.x) {
    int n = i / K, k = i - n * K;
    wt[i] = (_Float16)w[k * N + n];
  }
}

// ---- fused LN1 + shifted-window attention + proj + residual ------------------
__global__ __launch_bounds__(256) void k_attn(
    const float* __restrict__ x, const float* __restrict__ qkv_b,
    const float* __restrict__ proj_b, const float* __restrict__ rpb,
    const float* __restrict__ n1g, const float* __restrict__ n1b,
    const _Float16* __restrict__ qkvWt, const _Float16* __restrict__ projWt,
    float* __restrict__ out) {
  __shared__ _Float16 xt[64 * CC];       // 16 KB  LN'd window tokens (padded to 64 rows)
  __shared__ _Float16 qkvs[64 * 256];    // 32 KB  Q|K f16 (token-major)
  __shared__ _Float16 vts[CC * 64];      // 16 KB  V transposed (channel-major)
  __shared__ float    ss[64 * 64];       // 16 KB  scores (per head)
  __shared__ _Float16 ps[64 * 64];       //  8 KB  softmax probs
  __shared__ float    os[64 * CC];       // 32 KB  attention output (pre-proj)

  const int tid  = threadIdx.x;
  const int lane = tid & 31;
  const int wv   = __builtin_amdgcn_readfirstlane(tid >> 5);  // wave-uniform -> SGPR
  const int b    = blockIdx.x >> 6;
  const int w    = blockIdx.x & 63;
  const int wh   = w >> 3, ww = w & 7;
  const float scale = 0.17677669529663687f;  // 1/sqrt(32)

  // zero pad rows 49..63
  for (int i = tid; i < (64 - NTOK) * CC; i += 256) xt[NTOK * CC + i] = (_Float16)0.f;

  // gather (roll by -3) + LayerNorm1, one token per wave iteration
  for (int i = wv; i < NTOK; i += 8) {
    int yi = i / 7, xi = i - yi * 7;
    int sr = (wh * 7 + yi + SHFT) % HW;
    int sc = (ww * 7 + xi + SHFT) % HW;
    const float* row = x + ((size_t)b * (HW * HW) + (size_t)(sr * HW + sc)) * CC;
    float4 v = *(const float4*)(row + lane * 4);
    float mu = wred_sum(v.x + v.y + v.z + v.w) * (1.f / 128.f);
    float d0 = v.x - mu, d1 = v.y - mu, d2 = v.z - mu, d3 = v.w - mu;
    float var = wred_sum(d0 * d0 + d1 * d1 + d2 * d2 + d3 * d3) * (1.f / 128.f);
    float rs = rsqrtf(var + LN_EPS);
    int c = lane * 4;
    _Float16* dst = xt + i * CC + c;
    dst[0] = (_Float16)(d0 * rs * n1g[c + 0] + n1b[c + 0]);
    dst[1] = (_Float16)(d1 * rs * n1g[c + 1] + n1b[c + 1]);
    dst[2] = (_Float16)(d2 * rs * n1g[c + 2] + n1b[c + 2]);
    dst[3] = (_Float16)(d3 * rs * n1g[c + 3] + n1b[c + 3]);
  }
  __syncthreads();

  // QKV GEMM: (64x128) @ (128x384), 96 tiles over 8 waves (12 counted iters).
  // Region (Q/K/V) selected by scalar tile index nt -> s_cbranch.
  for (int tt = 0; tt < 12; ++tt) {
    int t = wv + tt * 8;
    int mt = t & 3, nt = t >> 2;
    v8f acc = {};
#pragma unroll
    for (int kk = 0; kk < 4; ++kk) {
      v16h a  = ld_a(xt, CC, mt * 16, kk * 32, lane);
      v16h bm = ld_a(qkvWt, CC, nt * 16, kk * 32, lane);
      acc = WMMA_F16(a, bm, acc);
    }
    int n = nt * 16 + (lane & 15);
    float bias = qkv_b[n];
    int mbase = mt * 16 + ((lane >> 4) << 3);
    if (nt < 8) {  // Q (pre-scaled)
#pragma unroll
      for (int r = 0; r < 8; ++r)
        qkvs[(mbase + r) * 256 + n] = (_Float16)((acc[r] + bias) * scale);
    } else if (nt < 16) {  // K
#pragma unroll
      for (int r = 0; r < 8; ++r)
        qkvs[(mbase + r) * 256 + n] = (_Float16)(acc[r] + bias);
    } else {  // V -> transposed, 8 consecutive halves per lane (one b128 store)
      _Float16* dst = vts + (n - 256) * 64 + mbase;
#pragma unroll
      for (int r = 0; r < 8; ++r) dst[r] = (_Float16)(acc[r] + bias);
    }
  }
  __syncthreads();

  // per-head attention
  for (int h = 0; h < NHEAD; ++h) {
    // S = Q K^T + rel_pos_bias + shift_mask : 16 tiles, 2 per wave
#pragma unroll
    for (int tt = 0; tt < 2; ++tt) {
      int t = wv * 2 + tt;
      int it = t >> 2, jt = t & 3;
      v16h a  = ld_a(qkvs, 256, it * 16, h * 32, lane);          // Q rows
      v16h bm = ld_a(qkvs, 256, jt * 16, 128 + h * 32, lane);    // K rows = cols of K^T
      v8f acc = {};
      acc = WMMA_F16(a, bm, acc);
      int j = jt * 16 + (lane & 15);
      int mbase = it * 16 + ((lane >> 4) << 3);
      int yj = j / 7, xj = j - yj * 7;
      int rhj = (wh == 7) ? (yj < 4 ? 1 : 2) : 0;
      int rwj = (ww == 7) ? (xj < 4 ? 1 : 2) : 0;
      int labj = rhj * 3 + rwj;
#pragma unroll
      for (int r = 0; r < 8; ++r) {
        int i = mbase + r;
        float val;
        if (j >= NTOK) {
          val = -1e30f;
        } else if (i >= NTOK) {
          val = 0.f;
        } else {
          int yi = i / 7, xi = i - yi * 7;
          int ridx = (yi - yj + 6) * 13 + (xi - xj + 6);
          float bias = rpb[ridx * NHEAD + h];
          int rhi = (wh == 7) ? (yi < 4 ? 1 : 2) : 0;
          int rwi = (ww == 7) ? (xi < 4 ? 1 : 2) : 0;
          float msk = ((rhi * 3 + rwi) != labj) ? -100.f : 0.f;
          val = acc[r] + bias + msk;
        }
        ss[i * 64 + j] = val;
      }
    }
    __syncthreads();

    // softmax: 4 lanes per row, 16 cols each (vectorized row access)
    {
      int row = tid >> 2, seg = tid & 3;
      const float4* srow = (const float4*)(ss + row * 64 + seg * 16);
      float4 f0 = srow[0], f1 = srow[1], f2 = srow[2], f3 = srow[3];
      float e[16] = {f0.x, f0.y, f0.z, f0.w, f1.x, f1.y, f1.z, f1.w,
                     f2.x, f2.y, f2.z, f2.w, f3.x, f3.y, f3.z, f3.w};
      float mx = -1e30f;
#pragma unroll
      for (int q = 0; q < 16; ++q) mx = fmaxf(mx, e[q]);
      mx = fmaxf(mx, __shfl_xor(mx, 1, 32));
      mx = fmaxf(mx, __shfl_xor(mx, 2, 32));
      float sm = 0.f;
#pragma unroll
      for (int q = 0; q < 16; ++q) { e[q] = __expf(e[q] - mx); sm += e[q]; }
      sm += __shfl_xor(sm, 1, 32);
      sm += __shfl_xor(sm, 2, 32);
      float inv = 1.f / sm;
      v8h p0, p1;
#pragma unroll
      for (int q = 0; q < 8; ++q) { p0[q] = (_Float16)(e[q] * inv); p1[q] = (_Float16)(e[q + 8] * inv); }
      v8h* prow = (v8h*)(ps + row * 64 + seg * 16);
      prow[0] = p0;
      prow[1] = p1;
    }
    __syncthreads();

    // O_h = P @ V : 8 tiles, 1 per wave (B-operand from transposed V -> contiguous)
    {
      int mt = wv >> 1, nt = wv & 1;
      v8f acc = {};
#pragma unroll
      for (int kk = 0; kk < 2; ++kk) {
        v16h a  = ld_a(ps, 64, mt * 16, kk * 32, lane);
        v16h bm = ld_a(vts, 64, h * 32 + nt * 16, kk * 32, lane);
        acc = WMMA_F16(a, bm, acc);
      }
      int n = h * 32 + nt * 16 + (lane & 15);
      int mbase = mt * 16 + ((lane >> 4) << 3);
#pragma unroll
      for (int r = 0; r < 8; ++r) os[(mbase + r) * CC + n] = acc[r];
    }
    __syncthreads();
  }

  // proj: (64x128)@(128x128) + bias, scatter with inverse roll, add residual
  for (int tt = 0; tt < 4; ++tt) {
    int t = wv + tt * 8;
    int mt = t & 3, nt = t >> 2;
    v8f acc = {};
#pragma unroll
    for (int kk = 0; kk < 4; ++kk) {
      v16h a  = ld_a_f32(os, CC, mt * 16, kk * 32, lane);
      v16h bm = ld_a(projWt, CC, nt * 16, kk * 32, lane);
      acc = WMMA_F16(a, bm, acc);
    }
    int n = nt * 16 + (lane & 15);
    float pb = proj_b[n];
    int mbase = mt * 16 + ((lane >> 4) << 3);
#pragma unroll
    for (int r = 0; r < 8; ++r) {
      int i = mbase + r;
      if (i < NTOK) {
        int yi = i / 7, xi = i - yi * 7;
        int sr = (wh * 7 + yi + SHFT) % HW;
        int sc = (ww * 7 + xi + SHFT) % HW;
        size_t idx = ((size_t)b * (HW * HW) + (size_t)(sr * HW + sc)) * CC + n;
        out[idx] = x[idx] + acc[r] + pb;
      }
    }
  }
}

// ---- fused LN2 + MLP (fc1 + GELU + fc2) + residual ---------------------------
__global__ __launch_bounds__(256) void k_mlp(
    const float* __restrict__ n2g, const float* __restrict__ n2b,
    const _Float16* __restrict__ fc1Wt, const float* __restrict__ fc1b,
    const _Float16* __restrict__ fc2Wt, const float* __restrict__ fc2b,
    float* __restrict__ out) {
  __shared__ _Float16 h2[64 * CC];    // 16 KB
  __shared__ _Float16 a1[64 * HID];   // 64 KB
  const int tid = threadIdx.x, lane = tid & 31;
  const int wv = __builtin_amdgcn_readfirstlane(tid >> 5);  // wave-uniform -> SGPR
  const size_t r0 = (size_t)blockIdx.x * 64;

  // LayerNorm2 on 64 rows of current out (= x + attn)
  for (int i = wv; i < 64; i += 8) {
    const float* row = out + (r0 + i) * CC;
    float4 v = *(const float4*)(row + lane * 4);
    float mu = wred_sum(v.x + v.y + v.z + v.w) * (1.f / 128.f);
    float d0 = v.x - mu, d1 = v.y - mu, d2 = v.z - mu, d3 = v.w - mu;
    float var = wred_sum(d0 * d0 + d1 * d1 + d2 * d2 + d3 * d3) * (1.f / 128.f);
    float rs = rsqrtf(var + LN_EPS);
    int c = lane * 4;
    _Float16* dst = h2 + i * CC + c;
    dst[0] = (_Float16)(d0 * rs * n2g[c + 0] + n2b[c + 0]);
    dst[1] = (_Float16)(d1 * rs * n2g[c + 1] + n2b[c + 1]);
    dst[2] = (_Float16)(d2 * rs * n2g[c + 2] + n2b[c + 2]);
    dst[3] = (_Float16)(d3 * rs * n2g[c + 3] + n2b[c + 3]);
  }
  __syncthreads();

  // fc1 + GELU(exact): (64x128)@(128x512), 128 tiles / 8 waves
  for (int tt = 0; tt < 16; ++tt) {
    int t = wv + tt * 8;
    int mt = t & 3, nt = t >> 2;
    v8f acc = {};
#pragma unroll
    for (int kk = 0; kk < 4; ++kk) {
      v16h a  = ld_a(h2, CC, mt * 16, kk * 32, lane);
      v16h bm = ld_a(fc1Wt, CC, nt * 16, kk * 32, lane);
      acc = WMMA_F16(a, bm, acc);
    }
    int n = nt * 16 + (lane & 15);
    float bb = fc1b[n];
    int mbase = mt * 16 + ((lane >> 4) << 3);
#pragma unroll
    for (int r = 0; r < 8; ++r) {
      float vv = acc[r] + bb;
      vv = 0.5f * vv * (1.f + erff(vv * 0.7071067811865475f));
      a1[(mbase + r) * HID + n] = (_Float16)vv;
    }
  }
  __syncthreads();

  // fc2 + residual: (64x512)@(512x128), 32 tiles / 8 waves
  for (int tt = 0; tt < 4; ++tt) {
    int t = wv + tt * 8;
    int mt = t & 3, nt = t >> 2;
    v8f acc = {};
#pragma unroll
    for (int kk = 0; kk < 16; ++kk) {
      v16h a  = ld_a(a1, HID, mt * 16, kk * 32, lane);
      v16h bm = ld_a(fc2Wt, HID, nt * 16, kk * 32, lane);
      acc = WMMA_F16(a, bm, acc);
    }
    int n = nt * 16 + (lane & 15);
    float bb = fc2b[n];
    int mbase = mt * 16 + ((lane >> 4) << 3);
#pragma unroll
    for (int r = 0; r < 8; ++r) {
      size_t row = r0 + (size_t)(mbase + r);
      out[row * CC + n] += acc[r] + bb;
    }
  }
}

extern "C" void kernel_launch(void* const* d_in, const int* in_sizes, int n_in,
                              void* d_out, int out_size, void* d_ws, size_t ws_size,
                              hipStream_t stream) {
  (void)in_sizes; (void)n_in; (void)out_size; (void)ws_size;
  const float* x      = (const float*)d_in[0];
  const float* qkv_w  = (const float*)d_in[1];
  const float* qkv_b  = (const float*)d_in[2];
  const float* proj_w = (const float*)d_in[3];
  const float* proj_b = (const float*)d_in[4];
  const float* rpb    = (const float*)d_in[5];
  const float* n1g    = (const float*)d_in[6];
  const float* n1b    = (const float*)d_in[7];
  const float* n2g    = (const float*)d_in[8];
  const float* n2b    = (const float*)d_in[9];
  const float* fc1w   = (const float*)d_in[10];
  const float* fc1b   = (const float*)d_in[11];
  const float* fc2w   = (const float*)d_in[12];
  const float* fc2b   = (const float*)d_in[13];
  float* out = (float*)d_out;

  _Float16* qkvWt  = (_Float16*)d_ws;          // [384][128]
  _Float16* projWt = qkvWt + 384 * 128;        // [128][128]
  _Float16* fc1Wt  = projWt + 128 * 128;       // [512][128]
  _Float16* fc2Wt  = fc1Wt + 512 * 128;        // [128][512]

  k_wprep<<<96, 256, 0, stream>>>(qkv_w, qkvWt, 128, 384);
  k_wprep<<<32, 256, 0, stream>>>(proj_w, projWt, 128, 128);
  k_wprep<<<128, 256, 0, stream>>>(fc1w, fc1Wt, 128, 512);
  k_wprep<<<128, 256, 0, stream>>>(fc2w, fc2Wt, 512, 128);

  k_attn<<<64 * 64, 256, 0, stream>>>(x, qkv_b, proj_b, rpb, n1g, n1b, qkvWt, projWt, out);
  k_mlp<<<3136, 256, 0, stream>>>(n2g, n2b, fc1Wt, fc1b, fc2Wt, fc2b, out);
}
